// MultiheadAttention_20839181320779
// MI455X (gfx1250) — matrix operand order repaired and checked by reading, not compiled, not run on hardware
//
#include <hip/hip_runtime.h>
#include <hip/hip_fp16.h>

typedef __attribute__((ext_vector_type(16))) _Float16 v16h;
typedef __attribute__((ext_vector_type(8)))  _Float16 v8h;
typedef __attribute__((ext_vector_type(4)))  _Float16 v4h;
typedef __attribute__((ext_vector_type(2)))  _Float16 v2h;
typedef __attribute__((ext_vector_type(8)))  float    v8f;

#define B_   2
#define N_   2048
#define D_   1024
#define H_   16
#define DH_  64
#define SCALE_ 0.125f  // DH^-0.5

#define WMMA_F16(A, Bf, C) \
  __builtin_amdgcn_wmma_f32_16x16x32_f16(false, (A), false, (Bf), (short)0, (C), false, false)

#define SHUF16(lo, hi) \
  __builtin_shufflevector((lo), (hi), 0, 1, 2, 3, 4, 5, 6, 7, 8, 9, 10, 11, 12, 13, 14, 15)

// ---------------------------------------------------------------------------
// Kernel 1: QKV projections.  Y = X @ W + b, output f16 in [b][h][n][dh].
// grid = (D/128, B*N/64, 3), block = 256 (8 waves; each wave owns a 32x32 C
// tile = 2x2 WMMA register block).  k-step 32.
// LDS: As[m][k] row-major; Bt[n][k] transposed at store time (packed b64
// stores, 4 k-values per store).  All fragments = 2x ds_load_b128.
// ---------------------------------------------------------------------------
__global__ __launch_bounds__(256)
void qkv_proj_kernel(const float* __restrict__ q, const float* __restrict__ k,
                     const float* __restrict__ v,
                     const float* __restrict__ Wq, const float* __restrict__ Wk,
                     const float* __restrict__ Wv,
                     const float* __restrict__ bq, const float* __restrict__ bk,
                     const float* __restrict__ bv,
                     _Float16* __restrict__ Qh, _Float16* __restrict__ Kh,
                     _Float16* __restrict__ Vh) {
  const int z = blockIdx.z;
  const float* X    = (z == 0) ? q  : (z == 1) ? k  : v;
  const float* W    = (z == 0) ? Wq : (z == 1) ? Wk : Wv;
  const float* bias = (z == 0) ? bq : (z == 1) ? bk : bv;
  _Float16* out     = (z == 0) ? Qh : (z == 1) ? Kh : Vh;

  const int tid  = threadIdx.x;
  const int lane = tid & 31;
  const int w    = tid >> 5;
  const int wm   = w >> 2;       // 0..1 -> 32-row group
  const int wn   = w & 3;        // 0..3 -> 32-col group
  const int nl   = lane & 15;
  const int g    = lane >> 4;

  const int row0 = blockIdx.y * 64;
  const int col0 = blockIdx.x * 128;

  __shared__ _Float16 As[64][40];    // [m][k], stride 80 B (16-aligned)
  __shared__ _Float16 Bt[128][40];   // [n][k], transposed W tile

  v8f acc[2][2] = {};

  for (int k0 = 0; k0 < D_; k0 += 32) {
    {  // A tile 64 x 32 fp32 -> f16, vectorized
      int r = tid >> 2, c = (tid & 3) * 8;
      const float4* src = (const float4*)(X + (size_t)(row0 + r) * D_ + k0 + c);
      float4 f0 = src[0], f1 = src[1];
      v8h hv = {(_Float16)f0.x, (_Float16)f0.y, (_Float16)f0.z, (_Float16)f0.w,
                (_Float16)f1.x, (_Float16)f1.y, (_Float16)f1.z, (_Float16)f1.w};
      *(v8h*)&As[r][c] = hv;
      if (k0 + 32 < D_)
        __builtin_prefetch(X + (size_t)(row0 + r) * D_ + k0 + 32 + c, 0, 3);
    }
    {  // W tile 32(k) x 128(n): coalesced reads of 4 k-rows, packed b64
       // transposed stores (4 contiguous k per store in Bt[n][k])
      int kr = (tid >> 5) * 4;       // 0,4,...,28
      int c  = (tid & 31) * 4;       // 0..124
      const float* Wb = W + (size_t)(k0 + kr) * D_ + col0 + c;
      float4 f0 = *(const float4*)(Wb);
      float4 f1 = *(const float4*)(Wb + D_);
      float4 f2 = *(const float4*)(Wb + 2 * D_);
      float4 f3 = *(const float4*)(Wb + 3 * D_);
      float a0[4] = {f0.x, f0.y, f0.z, f0.w};
      float a1[4] = {f1.x, f1.y, f1.z, f1.w};
      float a2[4] = {f2.x, f2.y, f2.z, f2.w};
      float a3[4] = {f3.x, f3.y, f3.z, f3.w};
#pragma unroll
      for (int i = 0; i < 4; ++i) {
        v4h hv = {(_Float16)a0[i], (_Float16)a1[i], (_Float16)a2[i], (_Float16)a3[i]};
        *(v4h*)&Bt[c + i][kr] = hv;
      }
      if (k0 + 32 < D_)
        __builtin_prefetch(Wb + 32 * D_, 0, 3);
    }
    __syncthreads();

    v16h av[2], bv2[2];
    {
      const int ar0 = wm * 32 + nl;
      av[0] = SHUF16(*(const v8h*)&As[ar0][g * 8],
                     *(const v8h*)&As[ar0][16 + g * 8]);
      av[1] = SHUF16(*(const v8h*)&As[ar0 + 16][g * 8],
                     *(const v8h*)&As[ar0 + 16][16 + g * 8]);
      const int br0 = wn * 32 + nl;
      bv2[0] = SHUF16(*(const v8h*)&Bt[br0][g * 8],
                      *(const v8h*)&Bt[br0][16 + g * 8]);
      bv2[1] = SHUF16(*(const v8h*)&Bt[br0 + 16][g * 8],
                      *(const v8h*)&Bt[br0 + 16][16 + g * 8]);
    }
#pragma unroll
    for (int sm = 0; sm < 2; ++sm)
#pragma unroll
      for (int sn = 0; sn < 2; ++sn)
        acc[sm][sn] = WMMA_F16(av[sm], bv2[sn], acc[sm][sn]);
    __syncthreads();
  }

#pragma unroll
  for (int sm = 0; sm < 2; ++sm) {
#pragma unroll
    for (int sn = 0; sn < 2; ++sn) {
      const int cc = col0 + wn * 32 + sn * 16 + nl;
      const float bb = bias[cc];
      const int hh = cc >> 6, dh = cc & 63;
#pragma unroll
      for (int r = 0; r < 8; ++r) {
        int rr = row0 + wm * 32 + sm * 16 + r + 8 * g;
        int bi = rr >> 11;           // / N_
        int nn = rr & (N_ - 1);
        out[((size_t)(bi * H_ + hh) * N_ + nn) * DH_ + dh] =
            (_Float16)(acc[sm][sn][r] + bb);
      }
    }
  }
}

// ---------------------------------------------------------------------------
// Kernel 2: flash attention.  One block per (b*h, 64-query tile).
// 8 waves: wq = w&3 (16-row query sub-tile), wk = w>>2 (key/dh column split).
// ---------------------------------------------------------------------------
__global__ __launch_bounds__(256)
void flash_attn_kernel(const _Float16* __restrict__ Qh,
                       const _Float16* __restrict__ Kh,
                       const _Float16* __restrict__ Vh,
                       const unsigned char* __restrict__ mask,
                       _Float16* __restrict__ Xout) {
  const int tid  = threadIdx.x;
  const int lane = tid & 31;
  const int w    = tid >> 5;
  const int wq   = w & 3;
  const int wk   = w >> 2;       // 0..1
  const int nl   = lane & 15;
  const int g    = lane >> 4;
  const int rb   = wq * 16 + 8 * g;   // base row (per-lane rows rb..rb+7)

  const int bh = blockIdx.y;
  const int b  = bh >> 4;
  const int h  = bh & 15;
  const int q0 = blockIdx.x * 64;

  const _Float16* Qb = Qh + (size_t)bh * N_ * DH_;
  const _Float16* Kb = Kh + (size_t)bh * N_ * DH_;
  const _Float16* Vb = Vh + (size_t)bh * N_ * DH_;
  const unsigned char* mb = mask + (size_t)b * N_ * N_;

  __shared__ _Float16 Qs[64][72];          // [qrow][d]
  __shared__ _Float16 Ks[32][72];          // [key][d]
  __shared__ _Float16 Vt[64][40];          // [dh][key]  (transposed V tile)
  __shared__ _Float16 Ps[64][40];          // [qrow][key]
  __shared__ unsigned char Ms[64][40];     // mask tile [qrow][key]
  __shared__ float m_run[64], l_run[64];
  __shared__ float red_max[2][64], red_sum[2][64];

  if (tid < 64) { m_run[tid] = -3.0e38f; l_run[tid] = 0.0f; }
  {  // load Q tile 64x64 (two b128 copies per thread)
    int r = tid >> 2, c = (tid & 3) * 16;
    const v8h* src = (const v8h*)(Qb + (size_t)(q0 + r) * DH_ + c);
    *(v8h*)&Qs[r][c]     = src[0];
    *(v8h*)&Qs[r][c + 8] = src[1];
  }
  __syncthreads();

  // persistent Q A-fragments (d = 0..31 and 32..63)
  v16h qa0, qa1;
  {
    const int qr = wq * 16 + nl;
    qa0 = SHUF16(*(const v8h*)&Qs[qr][g * 8],      *(const v8h*)&Qs[qr][16 + g * 8]);
    qa1 = SHUF16(*(const v8h*)&Qs[qr][32 + g * 8], *(const v8h*)&Qs[qr][48 + g * 8]);
  }

  v8f o0 = {}, o1 = {};

  for (int kt = 0; kt < N_; kt += 32) {
    {  // K tile 32x64 (b128 copy)
      int r = tid >> 3, c = (tid & 7) * 8;
      *(v8h*)&Ks[r][c] = *(const v8h*)(Kb + (size_t)(kt + r) * DH_ + c);
      if (kt + 32 < N_) {
        __builtin_prefetch(Kb + (size_t)(kt + 32 + r) * DH_ + c, 0, 3);
        __builtin_prefetch(Vb + (size_t)(kt + 32 + r) * DH_ + c, 0, 3);
      }
    }
    {  // V tile transposed: 2 keys x 4 dh per thread, packed b32 stores
      int r2 = (tid >> 4) * 2;     // 0,2,...,30
      int c4 = (tid & 15) * 4;     // 0..60
      v4h h0 = *(const v4h*)(Vb + (size_t)(kt + r2) * DH_ + c4);
      v4h h1 = *(const v4h*)(Vb + (size_t)(kt + r2 + 1) * DH_ + c4);
#pragma unroll
      for (int i = 0; i < 4; ++i) {
        v2h pv = {h0[i], h1[i]};
        *(v2h*)&Vt[c4 + i][r2] = pv;
      }
    }
    {  // mask tile (coalesced 8B loads)
      int mr = tid >> 2, mc = (tid & 3) * 8;
      *(uint2*)&Ms[mr][mc] =
          *(const uint2*)(mb + (size_t)(q0 + mr) * N_ + kt + mc);
    }
    __syncthreads();

    // S = Q K^T : 16 rows x 16 keys per wave, keys kt + wk*16 + nl
    v16h kb0, kb1;
    {
      const int kr = wk * 16 + nl;
      kb0 = SHUF16(*(const v8h*)&Ks[kr][g * 8],
                   *(const v8h*)&Ks[kr][16 + g * 8]);
      kb1 = SHUF16(*(const v8h*)&Ks[kr][32 + g * 8],
                   *(const v8h*)&Ks[kr][48 + g * 8]);
    }
    v8f s = {};
    s = WMMA_F16(qa0, kb0, s);
    s = WMMA_F16(qa1, kb1, s);

    // scale + mask (finite "-inf" keeps softmax NaN-free)
    float sv[8], pm[8], mold[8];
#pragma unroll
    for (int r = 0; r < 8; ++r) {
      float x = s[r] * SCALE_;
      if (Ms[rb + r][wk * 16 + nl]) x = -3.0e38f;
      sv[r] = x;
    }
    // row max over the 16 lanes sharing a row set (xor < 16 stays in half)
#pragma unroll
    for (int r = 0; r < 8; ++r) {
      float x = sv[r];
      x = fmaxf(x, __shfl_xor(x, 1, 32));
      x = fmaxf(x, __shfl_xor(x, 2, 32));
      x = fmaxf(x, __shfl_xor(x, 4, 32));
      x = fmaxf(x, __shfl_xor(x, 8, 32));
      pm[r] = x;
    }
#pragma unroll
    for (int r = 0; r < 8; ++r) mold[r] = m_run[rb + r];
    if (nl == 0) {
#pragma unroll
      for (int r = 0; r < 8; ++r) red_max[wk][rb + r] = pm[r];
    }
    __syncthreads();

    float mnew[8], fac[8], p[8], psum[8];
#pragma unroll
    for (int r = 0; r < 8; ++r) {
      float mm = fmaxf(red_max[0][rb + r], red_max[1][rb + r]);
      mm = fmaxf(mm, mold[r]);
      mnew[r] = mm;
      fac[r]  = __expf(mold[r] - mm);
      p[r]    = __expf(sv[r] - mm);
    }
#pragma unroll
    for (int r = 0; r < 8; ++r) {
      float x = p[r];
      x += __shfl_xor(x, 1, 32);
      x += __shfl_xor(x, 2, 32);
      x += __shfl_xor(x, 4, 32);
      x += __shfl_xor(x, 8, 32);
      psum[r] = x;
    }
    if (nl == 0) {
#pragma unroll
      for (int r = 0; r < 8; ++r) red_sum[wk][rb + r] = psum[r];
    }
    // rescale accumulators, stash P tile for the P*V matmul
#pragma unroll
    for (int r = 0; r < 8; ++r) {
      o0[r] *= fac[r];
      o1[r] *= fac[r];
      Ps[rb + r][wk * 16 + nl] = (_Float16)p[r];
    }
    __syncthreads();  // Ps + red_sum visible

    if (wk == 0 && nl == 0) {  // single-writer running-stat update
#pragma unroll
      for (int r = 0; r < 8; ++r) {
        int rr = rb + r;
        l_run[rr] = l_run[rr] * fac[r] + red_sum[0][rr] + red_sum[1][rr];
        m_run[rr] = mnew[r];
      }
    }

    // O += P * V : each wave owns dh columns [wk*32, wk*32+32)
    v16h pa, vb0, vb1;
    {
      const int pr = wq * 16 + nl;
      pa = SHUF16(*(const v8h*)&Ps[pr][g * 8],
                  *(const v8h*)&Ps[pr][16 + g * 8]);
      const int c0 = wk * 32 + nl;
      vb0 = SHUF16(*(const v8h*)&Vt[c0][g * 8],
                   *(const v8h*)&Vt[c0][16 + g * 8]);
      vb1 = SHUF16(*(const v8h*)&Vt[c0 + 16][g * 8],
                   *(const v8h*)&Vt[c0 + 16][16 + g * 8]);
    }
    o0 = WMMA_F16(pa, vb0, o0);
    o1 = WMMA_F16(pa, vb1, o1);
    __syncthreads();  // protect Ks/Vt/Ps/Ms + stats before next iteration
  }

  // normalize and write [b][n][h*64+dh] as f16
#pragma unroll
  for (int r = 0; r < 8; ++r) {
    float inv = 1.0f / l_run[rb + r];
    size_t o = ((size_t)b * N_ + (q0 + rb + r)) * D_ + h * DH_;
    Xout[o + wk * 32 + nl]      = (_Float16)(o0[r] * inv);
    Xout[o + wk * 32 + 16 + nl] = (_Float16)(o1[r] * inv);
  }
}

// ---------------------------------------------------------------------------
// Kernel 3: output projection.  out = Xa @ Wp + bp, fp32 out.
// Same 2x2-register-block structure as kernel 1 (A already f16).
// ---------------------------------------------------------------------------
__global__ __launch_bounds__(256)
void out_proj_kernel(const _Float16* __restrict__ Xin,
                     const float* __restrict__ Wp,
                     const float* __restrict__ bp,
                     float* __restrict__ Out) {
  const int tid  = threadIdx.x;
  const int lane = tid & 31;
  const int w    = tid >> 5;
  const int wm   = w >> 2, wn = w & 3;
  const int nl   = lane & 15, g = lane >> 4;

  const int row0 = blockIdx.y * 64;
  const int col0 = blockIdx.x * 128;

  __shared__ _Float16 As[64][40];
  __shared__ _Float16 Bt[128][40];

  v8f acc[2][2] = {};
  for (int k0 = 0; k0 < D_; k0 += 32) {
    {  // A tile 64x32 f16: straight b128 copy
      int r = tid >> 2, c = (tid & 3) * 8;
      *(v8h*)&As[r][c] = *(const v8h*)(Xin + (size_t)(row0 + r) * D_ + k0 + c);
    }
    {  // W tile transposed into Bt[n][k], packed b64 stores
      int kr = (tid >> 5) * 4;
      int c  = (tid & 31) * 4;
      const float* Wb = Wp + (size_t)(k0 + kr) * D_ + col0 + c;
      float4 f0 = *(const float4*)(Wb);
      float4 f1 = *(const float4*)(Wb + D_);
      float4 f2 = *(const float4*)(Wb + 2 * D_);
      float4 f3 = *(const float4*)(Wb + 3 * D_);
      float a0[4] = {f0.x, f0.y, f0.z, f0.w};
      float a1[4] = {f1.x, f1.y, f1.z, f1.w};
      float a2[4] = {f2.x, f2.y, f2.z, f2.w};
      float a3[4] = {f3.x, f3.y, f3.z, f3.w};
#pragma unroll
      for (int i = 0; i < 4; ++i) {
        v4h hv = {(_Float16)a0[i], (_Float16)a1[i], (_Float16)a2[i], (_Float16)a3[i]};
        *(v4h*)&Bt[c + i][kr] = hv;
      }
      if (k0 + 32 < D_)
        __builtin_prefetch(Wb + 32 * D_, 0, 3);
    }
    __syncthreads();

    v16h av[2], bv2[2];
    {
      const int ar0 = wm * 32 + nl;
      av[0] = SHUF16(*(const v8h*)&As[ar0][g * 8],
                     *(const v8h*)&As[ar0][16 + g * 8]);
      av[1] = SHUF16(*(const v8h*)&As[ar0 + 16][g * 8],
                     *(const v8h*)&As[ar0 + 16][16 + g * 8]);
      const int br0 = wn * 32 + nl;
      bv2[0] = SHUF16(*(const v8h*)&Bt[br0][g * 8],
                      *(const v8h*)&Bt[br0][16 + g * 8]);
      bv2[1] = SHUF16(*(const v8h*)&Bt[br0 + 16][g * 8],
                      *(const v8h*)&Bt[br0 + 16][16 + g * 8]);
    }
#pragma unroll
    for (int sm = 0; sm < 2; ++sm)
#pragma unroll
      for (int sn = 0; sn < 2; ++sn)
        acc[sm][sn] = WMMA_F16(av[sm], bv2[sn], acc[sm][sn]);
    __syncthreads();
  }

#pragma unroll
  for (int sm = 0; sm < 2; ++sm) {
#pragma unroll
    for (int sn = 0; sn < 2; ++sn) {
      const int cc = col0 + wn * 32 + sn * 16 + nl;
      const float bb = bp[cc];
#pragma unroll
      for (int r = 0; r < 8; ++r) {
        int rr = row0 + wm * 32 + sm * 16 + r + 8 * g;
        Out[(size_t)rr * D_ + cc] = acc[sm][sn][r] + bb;
      }
    }
  }
}

// ---------------------------------------------------------------------------
extern "C" void kernel_launch(void* const* d_in, const int* in_sizes, int n_in,
                              void* d_out, int out_size, void* d_ws, size_t ws_size,
                              hipStream_t stream) {
  const float* q    = (const float*)d_in[0];
  const float* k    = (const float*)d_in[1];
  const float* v    = (const float*)d_in[2];
  const unsigned char* mask = (const unsigned char*)d_in[3];  // jax bool -> 1 byte
  const float* Wq = (const float*)d_in[4];
  const float* bq = (const float*)d_in[5];
  const float* Wk = (const float*)d_in[6];
  const float* bk = (const float*)d_in[7];
  const float* Wv = (const float*)d_in[8];
  const float* bv = (const float*)d_in[9];
  const float* Wp = (const float*)d_in[10];
  const float* bp = (const float*)d_in[11];

  // workspace: 4 f16 buffers of B*N*D (Qh, Kh, Vh, attn-out) = 32 MB total
  _Float16* ws = (_Float16*)d_ws;
  const size_t per = (size_t)B_ * N_ * D_;
  _Float16* Qh = ws;
  _Float16* Kh = ws + per;
  _Float16* Vh = ws + 2 * per;
  _Float16* Xa = ws + 3 * per;

  dim3 blk(256);
  dim3 g1(D_ / 128, (B_ * N_) / 64, 3);
  qkv_proj_kernel<<<g1, blk, 0, stream>>>(q, k, v, Wq, Wk, Wv, bq, bk, bv, Qh, Kh, Vh);

  dim3 g2(N_ / 64, B_ * H_);
  flash_attn_kernel<<<g2, blk, 0, stream>>>(Qh, Kh, Vh, mask, Xa);

  dim3 g3(D_ / 128, (B_ * N_) / 64);
  out_proj_kernel<<<g3, blk, 0, stream>>>(Xa, Wp, bp, (float*)d_out);
}
